// attention_block_35244501631176
// MI455X (gfx1250) — matrix operand order repaired and checked by reading, not compiled, not run on hardware
//
#include <hip/hip_runtime.h>
#include <hip/hip_bf16.h>
#include <math.h>

typedef __attribute__((ext_vector_type(16))) _Float16 v16h;
typedef __attribute__((ext_vector_type(8)))  float    v8f;

#define WMMA16x16x32(a, b, c) \
  __builtin_amdgcn_wmma_f32_16x16x32_f16(false, (a), false, (b), (short)0, (c), false, false)

// ---------------------------------------------------------------------------
// WMMA fragment helpers (ISA 7.12.2 layouts, wave32)
// ---------------------------------------------------------------------------

// A fragment 16x32 f16 from LDS tile sA[rows][ldA] (row = M, col = K)
__device__ __forceinline__ v16h frag_a_lds(const _Float16* sA, int ldA, int mo, int lane) {
  int m  = mo + (lane & 15);
  int kb = ((lane >> 4) & 1) << 3;   // lanes 16..31 hold the K+8 group
  v16h a;
#pragma unroll
  for (int j = 0; j < 8; ++j) {
    int base = ((j >> 2) << 4) + kb + ((j & 3) << 1);
    a[2 * j]     = sA[m * ldA + base];
    a[2 * j + 1] = sA[m * ldA + base + 1];
  }
  return a;
}

// B fragment 32x16 f16 from TRANSPOSED LDS tile sBT[64 n][ldB k]:
// lane n holds K 0..7,16..23 (lanes<16) or 8..15,24..31 (lanes>=16)
// -> two 16-byte runs per lane (16 bytes apart) -> ds_load_b128 x2
__device__ __forceinline__ v16h frag_b_ldsT(const _Float16* sBT, int ldB, int no, int lane) {
  int n  = no + (lane & 15);
  int kb = ((lane >> 4) & 1) << 3;
  v16h b;
#pragma unroll
  for (int e = 0; e < 16; ++e) b[e] = sBT[n * ldB + ((e >> 3) << 4) + kb + (e & 7)];
  return b;
}

__device__ __forceinline__ void store_d(float* out, int ldout, int mo, int no, int lane, v8f c) {
  int n     = no + (lane & 15);
  int mbase = mo + (((lane >> 4) & 1) << 3);
#pragma unroll
  for (int r = 0; r < 8; ++r) out[(size_t)(mbase + r) * ldout + n] = c[r];
}

// bias + PReLU epilogue, f16 output, rows guarded by Mv
__device__ __forceinline__ void store_dh(_Float16* out, int ldout, int mo, int no, int lane,
                                         v8f c, const float* bias, float al, int Mv) {
  int n     = no + (lane & 15);
  int mbase = mo + (((lane >> 4) & 1) << 3);
#pragma unroll
  for (int r = 0; r < 8; ++r) {
    int m = mbase + r;
    if (m < Mv) {
      float v = c[r] + bias[m];
      v = v >= 0.f ? v : al * v;
      out[(size_t)m * ldout + n] = (_Float16)v;
    }
  }
}

// ---------------------------------------------------------------------------
// Generic dense WMMA GEMM: D[M,N] = A[M,K] x B[K,N]; N mult of 64, K mult 32,
// A allocated with ceil(M/64)*64 rows. 128 threads, 64x64 block tile.
// If Dh != null: Dh = f16 prelu(D + bias) for rows < Mv; else Df = f32 D.
// ---------------------------------------------------------------------------
__global__ __launch_bounds__(128) void k_gemm(const _Float16* __restrict__ A,
                                              const _Float16* __restrict__ B,
                                              float* __restrict__ Df,
                                              _Float16* __restrict__ Dh,
                                              const float* __restrict__ bias,
                                              const float* __restrict__ alpha,
                                              int N, int K, int Mv) {
  __shared__ __align__(16) _Float16 sA[64 * 40];
  __shared__ __align__(16) _Float16 sBT[64 * 40];
  const int m0 = blockIdx.x * 64;
  const int n0 = blockIdx.y * 64;
  const int tid = threadIdx.x, lane = tid & 31, wave = tid >> 5;
  const int mo = (wave & 1) * 32, no = (wave >> 1) * 32;
  v8f acc0 = {}, acc1 = {}, acc2 = {}, acc3 = {};
  for (int kk = 0; kk < K; kk += 32) {
    // A tile 64x32 halfs = 256 uint4; 2 per thread
#pragma unroll
    for (int v = tid; v < 256; v += 128) {
      int r = v >> 2, c8 = v & 3;
      *(uint4*)&sA[r * 40 + c8 * 8] =
          *(const uint4*)(A + (size_t)(m0 + r) * K + kk + c8 * 8);
    }
    // B tile 32x64 halfs, stored transposed (n-major) for vector fragment reads
    {
      int r = tid >> 2, c8 = tid & 3;
      uint4 t4 = *(const uint4*)(B + (size_t)(kk + r) * N + n0 + c8 * 8);
      const _Float16* tp = (const _Float16*)&t4;
#pragma unroll
      for (int j = 0; j < 8; ++j) sBT[(c8 * 8 + j) * 40 + r] = tp[j];
    }
    if (kk + 32 < K) {  // prefetch next K tile (global_prefetch_b8)
      __builtin_prefetch(A + (size_t)(m0 + (tid >> 1)) * K + kk + 32, 0, 1);
      __builtin_prefetch(B + (size_t)(kk + 32 + (tid >> 2)) * N + n0, 0, 1);
    }
    __syncthreads();
    v16h a0 = frag_a_lds(sA, 40, mo, lane);
    v16h a1 = frag_a_lds(sA, 40, mo + 16, lane);
    v16h b0 = frag_b_ldsT(sBT, 40, no, lane);
    v16h b1 = frag_b_ldsT(sBT, 40, no + 16, lane);
    acc0 = WMMA16x16x32(a0, b0, acc0);
    acc1 = WMMA16x16x32(a0, b1, acc1);
    acc2 = WMMA16x16x32(a1, b0, acc2);
    acc3 = WMMA16x16x32(a1, b1, acc3);
    __syncthreads();
  }
  if (Dh) {
    float al = alpha[0];
    store_dh(Dh, N, m0 + mo,      n0 + no,      lane, acc0, bias, al, Mv);
    store_dh(Dh, N, m0 + mo,      n0 + no + 16, lane, acc1, bias, al, Mv);
    store_dh(Dh, N, m0 + mo + 16, n0 + no,      lane, acc2, bias, al, Mv);
    store_dh(Dh, N, m0 + mo + 16, n0 + no + 16, lane, acc3, bias, al, Mv);
  } else {
    store_d(Df, N, m0 + mo,      n0 + no,      lane, acc0);
    store_d(Df, N, m0 + mo,      n0 + no + 16, lane, acc1);
    store_d(Df, N, m0 + mo + 16, n0 + no,      lane, acc2);
    store_d(Df, N, m0 + mo + 16, n0 + no + 16, lane, acc3);
  }
}

// ---------------------------------------------------------------------------
// Prep kernels
// ---------------------------------------------------------------------------

// f32 -> f16 copy
__global__ __launch_bounds__(256) void k_f32toh(const float* __restrict__ in,
                                                _Float16* __restrict__ out, int n) {
  int idx = blockIdx.x * 256 + threadIdx.x;
  if (idx < n) out[idx] = (_Float16)in[idx];
}

// ds_w[i] block [64][64][3][3] -> f16 [64, 576] (already matching (c,dy,dx) order)
__global__ __launch_bounds__(256) void k_w3h(const float* __restrict__ w,
                                             _Float16* __restrict__ wh) {
  int idx = blockIdx.x * 256 + threadIdx.x;
  if (idx < 64 * 576) wh[idx] = (_Float16)w[idx];
}

// im2col for 3x3 conv over output grid Hs x Hs reading x [64,48,48]:
// im3[(c,dy,dx), l] = x[c, ly+dy-pp, lx+dx-pp] (zero pad; l >= L -> 0)
__global__ __launch_bounds__(256) void k_im2col3(const float* __restrict__ x,
                                                 _Float16* __restrict__ im3,
                                                 int Hs, int pp, int Lpad) {
  int idx = blockIdx.x * 256 + threadIdx.x;
  if (idx >= 576 * Lpad) return;
  int kidx = idx / Lpad, l = idx - kidx * Lpad;
  int L = Hs * Hs;
  _Float16 v = (_Float16)0.f;
  if (l < L) {
    int c = kidx / 9, r = kidx - c * 9;
    int dy = r / 3, dx = r - dy * 3;
    int ly = l / Hs, lx = l - ly * Hs;
    int iy = ly + dy - pp, ix = lx + dx - pp;
    if (iy >= 0 && iy < 48 && ix >= 0 && ix < 48)
      v = (_Float16)x[(size_t)c * 2304 + iy * 48 + ix];
  }
  im3[idx] = v;
}

// out_f16 = prelu(W[Co,Ci] @ in_f16[Ci, stride inStride] + b), n < HW
__global__ __launch_bounds__(256) void k_c1x1_h(const _Float16* __restrict__ in, int inStride,
                                                const float* __restrict__ w,
                                                const float* __restrict__ bias,
                                                const float* __restrict__ alpha,
                                                _Float16* __restrict__ out,
                                                int Co, int Ci, int HW) {
  int idx = blockIdx.x * 256 + threadIdx.x;
  if (idx >= Co * HW) return;
  int o = idx / HW, n = idx - o * HW;
  float acc = bias[o];
  for (int c = 0; c < Ci; ++c) acc += w[o * Ci + c] * (float)in[(size_t)c * inStride + n];
  float al = alpha[0];
  acc = acc >= 0.f ? acc : al * acc;
  out[idx] = (_Float16)acc;
}

// inv_norm[l] = 1 / max(||ref patch at l||, 1e-4); ref [32, L] f16
__global__ __launch_bounds__(256) void k_invnorm(const _Float16* __restrict__ ref,
                                                 float* __restrict__ invn,
                                                 int Hs, int ksz, int pp) {
  int l = blockIdx.x * 256 + threadIdx.x;
  int L = Hs * Hs;
  if (l >= L) return;
  int ly = l / Hs, lx = l - ly * Hs;
  float s = 0.f;
  for (int c = 0; c < 32; ++c) {
    const _Float16* rp = ref + (size_t)c * L;
    for (int dy = 0; dy < ksz; ++dy) {
      int sy = ly + dy - pp;
      if (sy < 0 || sy >= Hs) continue;
      for (int dx = 0; dx < ksz; ++dx) {
        int sx = lx + dx - pp;
        if (sx < 0 || sx >= Hs) continue;
        float v = (float)rp[sy * Hs + sx];
        s += v * v;
      }
    }
  }
  invn[l] = 1.f / fmaxf(sqrtf(s), 1e-4f);
}

// refT[l, c2] = ref[c2, l] (zero for pad rows l >= L)
__global__ __launch_bounds__(256) void k_build_refT(const _Float16* __restrict__ ref,
                                                    _Float16* __restrict__ refT,
                                                    int L, int Lpad) {
  int idx = blockIdx.x * 256 + threadIdx.x;
  if (idx >= Lpad * 32) return;
  int l = idx >> 5, c2 = idx & 31;
  refT[idx] = (l < L) ? ref[(size_t)c2 * L + l] : (_Float16)0.f;
}

// embA[c, l(padded)] = emb[c, l] (zero for l >= L)
__global__ __launch_bounds__(256) void k_build_embA(const _Float16* __restrict__ emb,
                                                    _Float16* __restrict__ embA,
                                                    int L, int Lpad) {
  int idx = blockIdx.x * 256 + threadIdx.x;
  if (idx >= 64 * Lpad) return;
  int c = idx / Lpad, l = idx - c * Lpad;
  embA[idx] = (l < L) ? emb[(size_t)c * L + l] : (_Float16)0.f;
}

// ---------------------------------------------------------------------------
// Separable diagonal box filter:
//   out[l,m] = sum_{e=-p..p} in[l + e*sl, m + e*sm]   (valid taps only)
// axis 0: sl=1,  sm=1  (coords: l%Hs, m%48);  axis 1: sl=Hs, sm=48 (l/Hs, m/48)
// ---------------------------------------------------------------------------
__global__ __launch_bounds__(256) void k_box_f32(const float* __restrict__ in,
                                                 float* __restrict__ out,
                                                 const float* __restrict__ scale, // nullable
                                                 int Hs, int L, int Lpad, int pp, int axis) {
  int idx = blockIdx.x * 256 + threadIdx.x;
  if (idx >= Lpad * 2304) return;
  int l = idx / 2304, m = idx - l * 2304;
  if (l >= L) { out[idx] = 0.f; return; }
  int cl = axis ? (l / Hs) : (l % Hs);
  int cm = axis ? (m / 48) : (m % 48);
  int sl = axis ? Hs : 1;
  int sm = axis ? 48 : 1;
  float s = 0.f;
  for (int e = -pp; e <= pp; ++e) {
    if ((unsigned)(cl + e) < (unsigned)Hs && (unsigned)(cm + e) < 48u)
      s += in[(size_t)(l + e * sl) * 2304 + (m + e * sm)];
  }
  if (scale) s *= scale[l];
  out[idx] = s;
}

__global__ __launch_bounds__(256) void k_box_f16(const _Float16* __restrict__ in,
                                                 _Float16* __restrict__ out,
                                                 int Hs, int L, int Lpad, int pp, int axis) {
  int idx = blockIdx.x * 256 + threadIdx.x;
  if (idx >= Lpad * 2304) return;
  int l = idx / 2304, m = idx - l * 2304;
  if (l >= L) { out[idx] = (_Float16)0.f; return; }
  int cl = axis ? (l / Hs) : (l % Hs);
  int cm = axis ? (m / 48) : (m % 48);
  int sl = axis ? Hs : 1;
  int sm = axis ? 48 : 1;
  float s = 0.f;
  for (int e = -pp; e <= pp; ++e) {
    if ((unsigned)(cl + e) < (unsigned)Hs && (unsigned)(cm + e) < 48u)
      s += (float)in[(size_t)(l + e * sl) * 2304 + (m + e * sm)];
  }
  out[idx] = (_Float16)s;
}

// ---------------------------------------------------------------------------
// Column softmax over l < L (scale 10), write f16 probabilities (pad rows -> 0)
// ---------------------------------------------------------------------------
__global__ __launch_bounds__(256) void k_softmax(const float* __restrict__ Yi,
                                                 _Float16* __restrict__ P,
                                                 int L, int Lpad, int N) {
  __shared__ float red[256];
  const int n = blockIdx.x, t = threadIdx.x;
  float mx = -3.4e38f;
  for (int l = t; l < L; l += 256) mx = fmaxf(mx, Yi[(size_t)l * N + n]);
  red[t] = mx; __syncthreads();
  for (int s = 128; s > 0; s >>= 1) { if (t < s) red[t] = fmaxf(red[t], red[t + s]); __syncthreads(); }
  mx = red[0]; __syncthreads();
  float sum = 0.f;
  for (int l = t; l < L; l += 256) sum += __expf(10.f * (Yi[(size_t)l * N + n] - mx));
  red[t] = sum; __syncthreads();
  for (int s = 128; s > 0; s >>= 1) { if (t < s) red[t] += red[t + s]; __syncthreads(); }
  float inv = 1.f / red[0];
  for (int l = t; l < Lpad; l += 256) {
    float pv = (l < L) ? __expf(10.f * (Yi[(size_t)l * N + n] - mx)) * inv : 0.f;
    P[(size_t)l * N + n] = (_Float16)pv;
  }
}

// ---------------------------------------------------------------------------
// Fusion + channel attention + residual
// ---------------------------------------------------------------------------
__global__ __launch_bounds__(256) void k_fusion(const float* __restrict__ ybr,
                                                const float* __restrict__ x,
                                                const float* __restrict__ fw,
                                                const float* __restrict__ fb,
                                                float* __restrict__ fused) {
  int idx = blockIdx.x * 256 + threadIdx.x;
  if (idx >= 64 * 2304) return;
  int o = idx / 2304, n = idx - o * 2304;
  float acc = fb[o];
  for (int i = 0; i < 3; ++i)
    for (int c = 0; c < 64; ++c) {
      float v = ybr[((size_t)i * 64 + c) * 2304 + n] + 2.f * x[(size_t)c * 2304 + n];
      acc += fw[o * 192 + i * 64 + c] * v;
    }
  fused[idx] = acc;
}

__global__ __launch_bounds__(256) void k_camean(const float* __restrict__ fused,
                                                float* __restrict__ mv) {
  __shared__ float red[256];
  int c = blockIdx.x, t = threadIdx.x;
  float s = 0.f;
  for (int n = t; n < 2304; n += 256) s += fused[(size_t)c * 2304 + n];
  red[t] = s; __syncthreads();
  for (int st = 128; st > 0; st >>= 1) { if (t < st) red[t] += red[t + st]; __syncthreads(); }
  if (t == 0) mv[c] = red[0] * (1.f / 2304.f);
}

__global__ __launch_bounds__(64) void k_cavec(const float* __restrict__ mv,
                                              const float* __restrict__ w1,
                                              const float* __restrict__ b1,
                                              const float* __restrict__ w2,
                                              const float* __restrict__ b2,
                                              float* __restrict__ y2) {
  __shared__ float y1[4];
  int t = threadIdx.x;
  if (t < 4) {
    float a = b1[t];
    for (int c = 0; c < 64; ++c) a += w1[t * 64 + c] * mv[c];
    y1[t] = a > 0.f ? a : 0.f;
  }
  __syncthreads();
  if (t < 64) {
    float a = b2[t];
    for (int j = 0; j < 4; ++j) a += w2[t * 4 + j] * y1[j];
    y2[t] = 1.f / (1.f + __expf(-a));
  }
}

__global__ __launch_bounds__(256) void k_final(const float* __restrict__ x,
                                               const float* __restrict__ fused,
                                               const float* __restrict__ y2,
                                               float* __restrict__ out) {
  int idx = blockIdx.x * 256 + threadIdx.x;
  if (idx >= 64 * 2304) return;
  int c = idx / 2304;
  out[idx] = x[idx] + fused[idx] * y2[c];
}

// ---------------------------------------------------------------------------
// Host orchestration
// ---------------------------------------------------------------------------
extern "C" void kernel_launch(void* const* d_in, const int* in_sizes, int n_in,
                              void* d_out, int out_size, void* d_ws, size_t ws_size,
                              hipStream_t stream) {
  (void)in_sizes; (void)n_in; (void)out_size; (void)ws_size;
  const float* x     = (const float*)d_in[0];
  const float* ds_w  = (const float*)d_in[1];
  const float* ds_b  = (const float*)d_in[2];
  const float* ds_a  = (const float*)d_in[3];
  const float* ma_w  = (const float*)d_in[4];
  const float* ma_b  = (const float*)d_in[5];
  const float* ma_a  = (const float*)d_in[6];
  const float* m1_w  = (const float*)d_in[7];
  const float* m1_b  = (const float*)d_in[8];
  const float* m1_a  = (const float*)d_in[9];
  const float* m2_w  = (const float*)d_in[10];
  const float* m2_b  = (const float*)d_in[11];
  const float* m2_a  = (const float*)d_in[12];
  const float* fu_w  = (const float*)d_in[13];
  const float* fu_b  = (const float*)d_in[14];
  const float* ca_w1 = (const float*)d_in[15];
  const float* ca_b1 = (const float*)d_in[16];
  const float* ca_w2 = (const float*)d_in[17];
  const float* ca_b2 = (const float*)d_in[18];
  float* out = (float*)d_out;

  const int HW = 2304;       // 48*48
  const int LMAX = 52 * 52;  // k=7 branch
  const int LPADMAX = 2752;

  char* base = (char*)d_ws;
  size_t off = 0;
  auto carve = [&](size_t bytes) -> char* {
    char* p = base + off;
    off += (bytes + 255) & ~(size_t)255;
    return p;
  };
  _Float16* xh     = (_Float16*)carve(sizeof(_Float16) * 64 * HW);
  _Float16* w3h    = (_Float16*)carve(sizeof(_Float16) * 64 * 576);
  _Float16* im3    = (_Float16*)carve(sizeof(_Float16) * 576 * (size_t)LPADMAX);
  _Float16* s_h    = (_Float16*)carve(sizeof(_Float16) * 64 * (size_t)LPADMAX);
  _Float16* emb    = (_Float16*)carve(sizeof(_Float16) * 64 * LMAX);
  _Float16* refb   = (_Float16*)carve(sizeof(_Float16) * 32 * LMAX);
  _Float16* matchb = (_Float16*)carve(sizeof(_Float16) * 32 * HW);
  float*    invn   = (float*)carve(sizeof(float) * LMAX);
  _Float16* refT   = (_Float16*)carve(sizeof(_Float16) * (size_t)LPADMAX * 32);
  _Float16* embA   = (_Float16*)carve(sizeof(_Float16) * 64 * (size_t)LPADMAX);
  float*    bufA   = (float*)carve(sizeof(float) * (size_t)LPADMAX * HW);
  float*    bufB   = (float*)carve(sizeof(float) * (size_t)LPADMAX * HW);
  _Float16* pm     = (_Float16*)carve(sizeof(_Float16) * (size_t)LPADMAX * HW);
  float*    ybr    = (float*)carve(sizeof(float) * 3 * 64 * HW);
  float*    fused  = (float*)carve(sizeof(float) * 64 * HW);
  float*    mv     = (float*)carve(sizeof(float) * 64);
  float*    y2     = (float*)carve(sizeof(float) * 64);
  _Float16* r1h    = (_Float16*)bufB;  // f16 alias for probability box temp

  const int KS3[3] = {3, 5, 7};
  for (int b = 0; b < 2; ++b) {
    const float* xb = x + (size_t)b * 64 * HW;
    k_f32toh<<<(64 * HW + 255) / 256, 256, 0, stream>>>(xb, xh, 64 * HW);
    for (int i = 0; i < 3; ++i) {
      const int ksz = KS3[i], pp = (ksz - 1) / 2;
      const int Hs = 46 + 2 * pp;            // 48, 50, 52
      const int L = Hs * Hs;
      const int Lpad = ((L + 63) / 64) * 64; // 2304, 2560, 2752
      const int tot4 = Lpad * HW;

      // s = prelu(conv3x3(x)) as WMMA GEMM: [64 x 576] x [576 x Lpad] -> f16
      k_w3h<<<(64 * 576 + 255) / 256, 256, 0, stream>>>(ds_w + (size_t)i * 64 * 576, w3h);
      k_im2col3<<<(576 * Lpad + 255) / 256, 256, 0, stream>>>(xb, im3, Hs, pp, Lpad);
      {
        dim3 g(1, Lpad / 64);
        k_gemm<<<g, 128, 0, stream>>>(w3h, im3, nullptr, s_h, ds_b + i * 64, ds_a + i,
                                      Lpad, 576, 64);
      }
      // 1x1 convs (small) on VALU, f16 in/out
      k_c1x1_h<<<(64 * L + 255) / 256, 256, 0, stream>>>(
          s_h, Lpad, ma_w + (size_t)i * 64 * 64, ma_b + i * 64, ma_a + i, emb, 64, 64, L);
      k_c1x1_h<<<(32 * L + 255) / 256, 256, 0, stream>>>(
          s_h, Lpad, m2_w + (size_t)i * 32 * 64, m2_b + i * 32, m2_a + i, refb, 32, 64, L);
      k_c1x1_h<<<(32 * HW + 255) / 256, 256, 0, stream>>>(
          xh, HW, m1_w + (size_t)i * 32 * 64, m1_b + i * 32, m1_a + i, matchb, 32, 64, HW);
      k_invnorm<<<(L + 255) / 256, 256, 0, stream>>>(refb, invn, Hs, ksz, pp);
      k_build_refT<<<(Lpad * 32 + 255) / 256, 256, 0, stream>>>(refb, refT, L, Lpad);

      // R[Lpad, 2304] = refT[Lpad,32] x match[32,2304]  (K = 32)
      {
        dim3 g(Lpad / 64, HW / 64);
        k_gemm<<<g, 128, 0, stream>>>(refT, matchb, bufA, nullptr, nullptr, nullptr, HW, 32, Lpad);
      }
      // logits = invn .* Box(R): diagonal box, x then y (scaled on last pass)
      k_box_f32<<<(tot4 + 255) / 256, 256, 0, stream>>>(bufA, bufB, nullptr, Hs, L, Lpad, pp, 0);
      k_box_f32<<<(tot4 + 255) / 256, 256, 0, stream>>>(bufB, bufA, invn, Hs, L, Lpad, pp, 1);
      // softmax over l (scale 10) -> p (f16)
      k_softmax<<<HW, 256, 0, stream>>>(bufA, pm, L, Lpad, HW);
      // q = Box(p) (f16)
      k_box_f16<<<(tot4 + 255) / 256, 256, 0, stream>>>(pm, r1h, Hs, L, Lpad, pp, 0);
      k_box_f16<<<(tot4 + 255) / 256, 256, 0, stream>>>(r1h, pm, Hs, L, Lpad, pp, 1);
      // y_i[64, 2304] = embA[64, Lpad] x q[Lpad, 2304]  (K = Lpad)
      k_build_embA<<<(64 * Lpad + 255) / 256, 256, 0, stream>>>(emb, embA, L, Lpad);
      {
        dim3 g(1, HW / 64);
        k_gemm<<<g, 128, 0, stream>>>(embA, pm, ybr + (size_t)i * 64 * HW, nullptr,
                                      nullptr, nullptr, HW, Lpad, 64);
      }
    }
    k_fusion<<<(64 * HW + 255) / 256, 256, 0, stream>>>(ybr, xb, fu_w, fu_b, fused);
    k_camean<<<64, 256, 0, stream>>>(fused, mv);
    k_cavec<<<1, 64, 0, stream>>>(mv, ca_w1, ca_b1, ca_w2, ca_b2, y2);
    k_final<<<(64 * HW + 255) / 256, 256, 0, stream>>>(xb, fused, y2, out + (size_t)b * 64 * HW);
  }
}
